// MemoryEfficientAttention_24026047054193
// MI455X (gfx1250) — compile-verified, matrix operand order checked by reading, not run on hardware
//
#include <hip/hip_runtime.h>

typedef __attribute__((ext_vector_type(16))) __bf16 v16bf;
typedef __attribute__((ext_vector_type(8)))  __bf16 v8bf;
typedef __attribute__((ext_vector_type(8)))  float  v8f;

// ---------------------------------------------------------------- helpers ---

#if defined(__has_builtin)
#if __has_builtin(__builtin_amdgcn_cvt_pk_bf16_f32)
#define HAVE_PK_BF16 1
#endif
#endif

// pack two fp32 -> two bf16 in one dword (hw packed cvt when available)
__device__ __forceinline__ unsigned bfpack(float lo, float hi) {
#ifdef HAVE_PK_BF16
  auto t = __builtin_amdgcn_cvt_pk_bf16_f32(lo, hi);
  return __builtin_bit_cast(unsigned, t);
#else
  unsigned ulo = __builtin_bit_cast(unsigned, lo);
  unsigned uhi = __builtin_bit_cast(unsigned, hi);
  unsigned rlo = (ulo + 0x7FFFu + ((ulo >> 16) & 1u)) >> 16;
  unsigned rhi = (uhi + 0x7FFFu + ((uhi >> 16) & 1u)) >> 16;
  return (rhi << 16) | (rlo & 0xFFFFu);
#endif
}

__device__ __forceinline__ __bf16 f2bf(float f) {
#ifdef HAVE_PK_BF16
  unsigned u = bfpack(f, 0.0f);
  return __builtin_bit_cast(__bf16, (unsigned short)u);
#else
  unsigned u = __builtin_bit_cast(unsigned, f);
  unsigned r = (u + 0x7FFFu + ((u >> 16) & 1u)) >> 16;
  return __builtin_bit_cast(__bf16, (unsigned short)r);
#endif
}

// 16-bit A/B WMMA fragment from a row-major 16x32 tile (ISA 7.12.2).
// Lane l<16 covers row l; lane half h selects K runs h*8.. and 16+h*8..
// Each run is 8 contiguous bf16 = one 16B LDS/global load.
// Requires: base 16B-aligned, ld a multiple of 8 elements.
__device__ __forceinline__ v16bf frag_row16(const __bf16* base, int ld, int lane) {
  int h = lane >> 4, rr = lane & 15;
  const __bf16* p = base + rr * ld + h * 8;
  v8bf lo = *(const v8bf*)(p);       // k = h*8 .. h*8+7      -> frag elems 0..7
  v8bf hi = *(const v8bf*)(p + 16);  // k = 16+h*8 .. +7      -> frag elems 8..15
  return __builtin_shufflevector(lo, hi, 0, 1, 2, 3, 4, 5, 6, 7,
                                 8, 9, 10, 11, 12, 13, 14, 15);
}

__device__ __forceinline__ v8f wmma_bf16(v16bf a, v16bf b, v8f c) {
  return __builtin_amdgcn_wmma_f32_16x16x32_bf16(false, a, false, b,
                                                 (short)0, c, false, false);
}

// ------------------------------------------------- constants for this model --
#define BB 2
#define SS 2048
#define HH 1024
#define NHH 16
#define HDD 64
#define MROWS (BB * SS)     // 4096
#define SCALEF 0.125f       // 1/sqrt(64)

// ============================ Kernel 1: QKV GEMM =============================
// X[4096,1024] fp32 @ Wqkv[1024,3072] fp32 + b -> Q/K/V bf16 [B,NH,S,64]
// Block tile 256(M) x 64(N), K-step 32; 8 waves each own 32 rows.
__global__ __launch_bounds__(256) void qkv_gemm_kernel(
    const float* __restrict__ X, const float* __restrict__ W,
    const float* __restrict__ bias,
    __bf16* __restrict__ Q, __bf16* __restrict__ K, __bf16* __restrict__ V) {
  __shared__ alignas(16) __bf16 As[256][40];  // row-major [m][k]
  __shared__ alignas(16) __bf16 Bs[64][40];   // column-major [n][k]

  int tid = threadIdx.x, wave = tid >> 5, lane = tid & 31;
  int m0 = blockIdx.x * 256;
  int n0 = blockIdx.y * 64;
  v8f acc[2][4] = {};

  // marching per-thread base pointers (constant strides become imm offsets)
  int ar = tid >> 3, ac = tid & 7;        // A: row band + 4-col quad
  int bn = tid & 63, bk = tid >> 6;       // B: column + k pair band
  const float* ax = X + (size_t)(m0 + ar) * HH + 4 * ac;
  const float* wx = W + (size_t)(2 * bk) * (3 * HH) + n0 + bn;

  for (int k0 = 0; k0 < HH; k0 += 32) {
    // A fill: float4 load -> 2 packed cvts -> one b64 LDS store (8/thread)
#pragma unroll
    for (int i = 0; i < 8; ++i) {
      float4 xv = *(const float4*)(ax + i * 32 * HH);
      unsigned long long pk =
          (unsigned long long)bfpack(xv.x, xv.y) |
          ((unsigned long long)bfpack(xv.z, xv.w) << 32);
      *(unsigned long long*)&As[ar + 32 * i][4 * ac] = pk;
    }
    // B fill (transpose to [n][k]): 1024 pairs along k
#pragma unroll
    for (int i = 0; i < 4; ++i) {
      float w0 = wx[i * 8 * (3 * HH)];
      float w1 = wx[i * 8 * (3 * HH) + (3 * HH)];
      *(unsigned*)&Bs[bn][2 * (bk + 4 * i)] = bfpack(w0, w1);
    }
    if (k0 + 32 < HH) {
      __builtin_prefetch(wx + 32 * (3 * HH), 0, 1);
    }
    ax += 32;
    wx += 32 * (3 * HH);
    __syncthreads();

    v16bf a0 = frag_row16(&As[wave * 32][0],      40, lane);
    v16bf a1 = frag_row16(&As[wave * 32 + 16][0], 40, lane);
#pragma unroll
    for (int nc = 0; nc < 4; ++nc) {
      v16bf b = frag_row16(&Bs[nc * 16][0], 40, lane);
      acc[0][nc] = wmma_bf16(a0, b, acc[0][nc]);
      acc[1][nc] = wmma_bf16(a1, b, acc[1][nc]);
    }
    __syncthreads();
  }

  // epilogue: bias + scatter into head-major Q/K/V (bf16)
  int half = lane >> 4, nl = lane & 15;
#pragma unroll
  for (int nc = 0; nc < 4; ++nc) {
    int n = n0 + nc * 16 + nl;            // [0,3072), fixed per nc
    float bv = bias[n];
    int which = n >> 10;
    int hn = n & 1023;
    int h = hn >> 6, hd = hn & 63;
    __bf16* dst0 = (which == 0) ? Q : ((which == 1) ? K : V);
#pragma unroll
    for (int mi = 0; mi < 2; ++mi) {
#pragma unroll
      for (int r = 0; r < 8; ++r) {
        int m = m0 + wave * 32 + mi * 16 + r + 8 * half;  // b*S + s
        int b = m >> 11, s = m & 2047;
        dst0[(((size_t)b * NHH + h) * SS + s) * HDD + hd] =
            f2bf(acc[mi][nc][r] + bv);
      }
    }
  }
}

// ======================= Kernel 2: flash attention ==========================
// per block: 128 query rows of one (b,h); streams K/V in 64-key tiles.
__global__ __launch_bounds__(256) void attn_kernel(
    const __bf16* __restrict__ Q, const __bf16* __restrict__ Kg,
    const __bf16* __restrict__ Vg, __bf16* __restrict__ Aout) {
  __shared__ alignas(16) __bf16 Ks[64][72];      // row-major [key][hd]
  __shared__ alignas(16) __bf16 Vcm[64][72];     // column-major [hd][key]
  __shared__ alignas(16) __bf16 Ps[8][16][72];   // per-wave P staging [q][key]

  int tid = threadIdx.x, wave = tid >> 5, lane = tid & 31;
  int half = lane >> 4, l16 = lane & 15;
  int qt = blockIdx.x;                 // 0..15
  int bh = blockIdx.y;                 // 0..31
  const __bf16* Qp = Q + (size_t)bh * SS * HDD;
  int qrow0 = qt * 128 + wave * 16;

  // marching fill pointers
  int kr = tid >> 3, kc8 = tid & 7;    // K: 64 rows x 8 b128-chunks
  int hdp = tid & 31, vkq = tid >> 5;  // V: 32 hd-pairs x 8 key-pair bands
  const __bf16* kb = Kg + (size_t)bh * SS * HDD + (size_t)kr * HDD + 8 * kc8;
  const __bf16* vb = Vg + (size_t)bh * SS * HDD + (size_t)(2 * vkq) * HDD + 2 * hdp;

  // Q A-fragments (hd 0..31, 32..63): 2 x global b128 each, in registers
  v16bf qa[2];
  {
    const __bf16* p = Qp + (size_t)(qrow0 + l16) * HDD + half * 8;
#pragma unroll
    for (int kk = 0; kk < 2; ++kk) {
      v8bf lo = *(const v8bf*)(p + kk * 32);
      v8bf hi = *(const v8bf*)(p + kk * 32 + 16);
      qa[kk] = __builtin_shufflevector(lo, hi, 0, 1, 2, 3, 4, 5, 6, 7,
                                       8, 9, 10, 11, 12, 13, 14, 15);
    }
  }

  v8f o[4] = {};
  float mrow[8], lrow[8];
#pragma unroll
  for (int r = 0; r < 8; ++r) { mrow[r] = -3.0e38f; lrow[r] = 0.0f; }

  for (int kt = 0; kt < SS; kt += 64) {
    // K tile: 2 x (global b128 -> ds b128) per thread into [key][hd]
#pragma unroll
    for (int i = 0; i < 2; ++i) {
      *(v8bf*)&Ks[kr + 32 * i][8 * kc8] = *(const v8bf*)(kb + i * 32 * HDD);
    }
    // V tile: 2x2 bf16 block transpose via v_perm into [hd][key]
#pragma unroll
    for (int i = 0; i < 4; ++i) {
      int kp = vkq + 8 * i;                               // key pair index
      unsigned va = *(const unsigned*)(vb + i * 16 * HDD);        // key 2kp
      unsigned vc = *(const unsigned*)(vb + i * 16 * HDD + HDD);  // key 2kp+1
      *(unsigned*)&Vcm[2 * hdp][2 * kp] =
          __builtin_amdgcn_perm(va, vc, 0x01000504u);  // lo halves
      *(unsigned*)&Vcm[2 * hdp + 1][2 * kp] =
          __builtin_amdgcn_perm(va, vc, 0x03020706u);  // hi halves
    }
    kb += 64 * HDD;
    vb += 64 * HDD;
    __syncthreads();

    // scores: 16q x 64k (4 column groups, two K=32 WMMAs chained over hd)
    v8f sc[4];
#pragma unroll
    for (int nc = 0; nc < 4; ++nc) {
      v16bf b0 = frag_row16(&Ks[nc * 16][0],  72, lane);
      v16bf b1 = frag_row16(&Ks[nc * 16][32], 72, lane);
      v8f z = {};
      sc[nc] = wmma_bf16(qa[0], b0, z);
      sc[nc] = wmma_bf16(qa[1], b1, sc[nc]);
    }

    // online softmax over 64 keys (row stats replicated across 16-lane half)
    float pv[4][8];
#pragma unroll
    for (int r = 0; r < 8; ++r) {
      float s0 = sc[0][r] * SCALEF, s1 = sc[1][r] * SCALEF;
      float s2 = sc[2][r] * SCALEF, s3 = sc[3][r] * SCALEF;
      float tmax = fmaxf(fmaxf(s0, s1), fmaxf(s2, s3));
#pragma unroll
      for (int sft = 1; sft < 16; sft <<= 1)
        tmax = fmaxf(tmax, __shfl_xor(tmax, sft, 32));
      float newm = fmaxf(mrow[r], tmax);
      float corr = __expf(mrow[r] - newm);
      mrow[r] = newm;
      pv[0][r] = __expf(s0 - newm);
      pv[1][r] = __expf(s1 - newm);
      pv[2][r] = __expf(s2 - newm);
      pv[3][r] = __expf(s3 - newm);
      float rs = (pv[0][r] + pv[1][r]) + (pv[2][r] + pv[3][r]);
#pragma unroll
      for (int sft = 1; sft < 16; sft <<= 1)
        rs += __shfl_xor(rs, sft, 32);
      lrow[r] = lrow[r] * corr + rs;
#pragma unroll
      for (int nc = 0; nc < 4; ++nc) o[nc][r] *= corr;
    }

    // C-layout P -> A-layout via per-wave LDS staging (scatter b16 stores,
    // wide b128 reads)
#pragma unroll
    for (int nc = 0; nc < 4; ++nc) {
#pragma unroll
      for (int r = 0; r < 8; ++r) {
        Ps[wave][r + 8 * half][nc * 16 + l16] = f2bf(pv[nc][r]);
      }
    }
    __syncthreads();

    v16bf pa0 = frag_row16(&Ps[wave][0][0],  72, lane);
    v16bf pa1 = frag_row16(&Ps[wave][0][32], 72, lane);
#pragma unroll
    for (int nc = 0; nc < 4; ++nc) {
      v16bf vb0 = frag_row16(&Vcm[nc * 16][0],  72, lane);
      v16bf vb1 = frag_row16(&Vcm[nc * 16][32], 72, lane);
      o[nc] = wmma_bf16(pa0, vb0, o[nc]);
      o[nc] = wmma_bf16(pa1, vb1, o[nc]);
    }
    __syncthreads();
  }

  // normalize and store to [b*S+s][H] bf16 (col = h*64 + hd)
  float inv[8];
#pragma unroll
  for (int r = 0; r < 8; ++r) inv[r] = 1.0f / lrow[r];
  int b = bh >> 4, h = bh & 15;
#pragma unroll
  for (int nc = 0; nc < 4; ++nc) {
#pragma unroll
    for (int r = 0; r < 8; ++r) {
      int qrow = qrow0 + r + 8 * half;
      int hd = nc * 16 + l16;
      Aout[((size_t)(b * SS + qrow)) * HH + h * HDD + hd] = f2bf(o[nc][r] * inv[r]);
    }
  }
}

// ========================== Kernel 3: output GEMM ===========================
// attn bf16 [4096,1024] @ w_out fp32 [1024,1024] + b_out -> out fp32
__global__ __launch_bounds__(256) void out_gemm_kernel(
    const __bf16* __restrict__ A, const float* __restrict__ W,
    const float* __restrict__ bias, float* __restrict__ out) {
  __shared__ alignas(16) __bf16 As[256][40];
  __shared__ alignas(16) __bf16 Bs[64][40];   // column-major [n][k]

  int tid = threadIdx.x, wave = tid >> 5, lane = tid & 31;
  int m0 = blockIdx.x * 256;
  int n0 = blockIdx.y * 64;
  v8f acc[2][4] = {};

  int ar = tid >> 2, ac8 = tid & 3;     // A: row band + b128 chunk
  int bn = tid & 63, bk = tid >> 6;
  const __bf16* ax = A + (size_t)(m0 + ar) * HH + 8 * ac8;
  const float*  wx = W + (size_t)(2 * bk) * HH + n0 + bn;

  for (int k0 = 0; k0 < HH; k0 += 32) {
    // A fill: bf16 source, 4 x (global b128 -> ds b128) per thread
#pragma unroll
    for (int i = 0; i < 4; ++i) {
      *(v8bf*)&As[ar + 64 * i][8 * ac8] = *(const v8bf*)(ax + i * 64 * HH);
    }
#pragma unroll
    for (int i = 0; i < 4; ++i) {
      float w0 = wx[i * 8 * HH];
      float w1 = wx[i * 8 * HH + HH];
      *(unsigned*)&Bs[bn][2 * (bk + 4 * i)] = bfpack(w0, w1);
    }
    if (k0 + 32 < HH) {
      __builtin_prefetch(wx + 32 * HH, 0, 1);
    }
    ax += 32;
    wx += 32 * HH;
    __syncthreads();

    v16bf a0 = frag_row16(&As[wave * 32][0],      40, lane);
    v16bf a1 = frag_row16(&As[wave * 32 + 16][0], 40, lane);
#pragma unroll
    for (int nc = 0; nc < 4; ++nc) {
      v16bf b = frag_row16(&Bs[nc * 16][0], 40, lane);
      acc[0][nc] = wmma_bf16(a0, b, acc[0][nc]);
      acc[1][nc] = wmma_bf16(a1, b, acc[1][nc]);
    }
    __syncthreads();
  }

  int half = lane >> 4, nl = lane & 15;
#pragma unroll
  for (int nc = 0; nc < 4; ++nc) {
    int n = n0 + nc * 16 + nl;
    float bv = bias[n];
#pragma unroll
    for (int mi = 0; mi < 2; ++mi) {
#pragma unroll
      for (int r = 0; r < 8; ++r) {
        int m = m0 + wave * 32 + mi * 16 + r + 8 * half;
        out[(size_t)m * HH + n] = acc[mi][nc][r] + bv;
      }
    }
  }
}

// ================================ launcher ==================================
extern "C" void kernel_launch(void* const* d_in, const int* in_sizes, int n_in,
                              void* d_out, int out_size, void* d_ws,
                              size_t ws_size, hipStream_t stream) {
  const float* x     = (const float*)d_in[0];
  const float* w_qkv = (const float*)d_in[1];
  const float* b_qkv = (const float*)d_in[2];
  const float* w_out = (const float*)d_in[3];
  const float* b_out = (const float*)d_in[4];
  float* out = (float*)d_out;

  char* ws = (char*)d_ws;
  const size_t headElems = (size_t)BB * NHH * SS * HDD;  // 4M bf16 = 8MB
  __bf16* Q     = (__bf16*)(ws);
  __bf16* K     = (__bf16*)(ws + headElems * 2);
  __bf16* V     = (__bf16*)(ws + headElems * 4);
  __bf16* Aattn = (__bf16*)(ws + headElems * 6);         // [4096,1024] bf16

  dim3 g1(MROWS / 256, (3 * HH) / 64);  // 16 x 48
  qkv_gemm_kernel<<<g1, 256, 0, stream>>>(x, w_qkv, b_qkv, Q, K, V);

  dim3 g2(SS / 128, BB * NHH);          // 16 x 32
  attn_kernel<<<g2, 256, 0, stream>>>(Q, K, V, Aattn);

  dim3 g3(MROWS / 256, HH / 64);        // 16 x 16
  out_gemm_kernel<<<g3, 256, 0, stream>>>(Aattn, w_out, b_out, out);
}
